// GraphNet_20641612825093
// MI455X (gfx1250) — matrix operand order repaired and checked by reading, not compiled, not run on hardware
//
#include <hip/hip_runtime.h>
#include <hip/hip_bf16.h>

typedef __attribute__((ext_vector_type(16))) _Float16 v16h;
typedef __attribute__((ext_vector_type(2)))  _Float16 v2h;
typedef __attribute__((ext_vector_type(8)))  float    v8f;

// Problem constants
#define BB   512
#define PP   30
#define NN   60
#define SS   14
#define NV   5
#define HID  64
#define DE   5
#define DO_  6
#define NT_  2
#define NPAIR (NN*(NN-1))   // 3540
#define NSV   (NN*NV)       // 300

// Padded f16 weight layout offsets (in halves) inside d_ws.
// Layer-1 weights: column-split remap (orig j<30 -> j ; j>=30 -> j+2).
// Layer-2 weights: input rows permuted to match h1 storage (k' holds logical (k'&3)*16+(k'>>2)).
// Layer-3 weights: input rows permuted to match h2 storage (k' holds logical (k'&1)*16+(k'>>1)).
#define OW1   0              // fr1  64x64 (split)
#define OW2   4096           // fr2  32x64 (perm4)
#define OW3   6144           // fr3  16x32 (perm2)
#define OW1s  6656           // fr1s 64x64 (split)
#define OW2s  10752          // fr2s 32x64 (perm4)
#define OW3s  12800          // fr3s 16x32 (perm2)
#define OWo1  13312          // fo1  64x64 (split)
#define OWo2  17408          // fo2  32x64 (perm4)
#define OWo3  19456          // fo3  16x32 (perm2)
#define NWH   19968          // total halves
// f32 bias region (floats), byte offset 40960 in d_ws
#define WSB_BYTE_OFF 40960
#define OB1   0
#define OB2   64
#define OB3   96
#define OB1s  112
#define OB2s  176
#define OB3s  208
#define OBo1  224
#define OBo2  288
#define OBo3  320

// ---------------- prep kernel ----------------
// mode: 0 = direct, 1 = split-at-30, 2 = perm4 (layer2), 3 = perm2 (layer3)
__device__ __forceinline__ void cvtW(const float* W, int out, int in, int outp, int inp,
                                     int mode, _Float16* dst, int tid, int nth) {
  for (int i = tid; i < outp * inp; i += nth) {
    int o = i / inp, k = i - o * inp;
    int j = -1;
    if (mode == 1) {
      if (k < 30) j = k;
      else if (k >= 32 && k < 32 + (in - 30)) j = 30 + (k - 32);
    } else if (mode == 2) {
      j = (k & 3) * 16 + (k >> 2);
      if (j >= in) j = -1;
    } else if (mode == 3) {
      j = (k & 1) * 16 + (k >> 1);
      if (j >= in) j = -1;
    } else {
      if (k < in) j = k;
    }
    float v = (o < out && j >= 0) ? W[o * in + j] : 0.0f;
    dst[i] = (_Float16)v;
  }
}
__device__ __forceinline__ void cvtB(const float* Bv, int n, int np, float* dst, int tid, int nth) {
  for (int i = tid; i < np; i += nth) dst[i] = (i < n) ? Bv[i] : 0.0f;
}

__global__ void prep_weights(const float* fr1_w, const float* fr1_b,
                             const float* fr2_w, const float* fr2_b,
                             const float* fr3_w, const float* fr3_b,
                             const float* fr1s_w, const float* fr1s_b,
                             const float* fr2s_w, const float* fr2s_b,
                             const float* fr3s_w, const float* fr3s_b,
                             const float* fo1_w, const float* fo1_b,
                             const float* fo2_w, const float* fo2_b,
                             const float* fo3_w, const float* fo3_b,
                             _Float16* wsH, float* wsB) {
  int tid = blockIdx.x * blockDim.x + threadIdx.x;
  int nth = gridDim.x * blockDim.x;
  cvtW(fr1_w,  HID, 2*PP,    HID, 64, 1, wsH + OW1,  tid, nth);
  cvtW(fr2_w,  32,  HID,     32,  64, 2, wsH + OW2,  tid, nth);
  cvtW(fr3_w,  DE,  32,      16,  32, 3, wsH + OW3,  tid, nth);
  cvtW(fr1s_w, HID, SS+PP,   HID, 64, 1, wsH + OW1s, tid, nth);
  cvtW(fr2s_w, 32,  HID,     32,  64, 2, wsH + OW2s, tid, nth);
  cvtW(fr3s_w, DE,  32,      16,  32, 3, wsH + OW3s, tid, nth);
  cvtW(fo1_w,  HID, PP+2*DE, HID, 64, 1, wsH + OWo1, tid, nth);
  cvtW(fo2_w,  32,  HID,     32,  64, 2, wsH + OWo2, tid, nth);
  cvtW(fo3_w,  DO_, 32,      16,  32, 3, wsH + OWo3, tid, nth);
  cvtB(fr1_b,  HID, 64, wsB + OB1,  tid, nth);
  cvtB(fr2_b,  32,  32, wsB + OB2,  tid, nth);
  cvtB(fr3_b,  DE,  16, wsB + OB3,  tid, nth);
  cvtB(fr1s_b, HID, 64, wsB + OB1s, tid, nth);
  cvtB(fr2s_b, 32,  32, wsB + OB2s, tid, nth);
  cvtB(fr3s_b, DE,  16, wsB + OB3s, tid, nth);
  cvtB(fo1_b,  HID, 64, wsB + OBo1, tid, nth);
  cvtB(fo2_b,  32,  32, wsB + OBo2, tid, nth);
  cvtB(fo3_b,  DO_, 16, wsB + OBo3, tid, nth);
}

// ---------------- WMMA fragment helpers (ISA 7.12.2 layouts, wave32) ----------------
// A: 16x32 f16 tile from row-major buffer (row stride ld halves).
__device__ __forceinline__ v16h loadA(const _Float16* h, int ld, int k0, int lane) {
  int m = lane & 15, hf = lane >> 4;
  v16h a;
#pragma unroll
  for (int v = 0; v < 8; ++v) {
    int kb = k0 + ((v >> 2) << 4) + (hf << 3) + ((v & 3) << 1);
    a[2 * v]     = h[m * ld + kb];
    a[2 * v + 1] = h[m * ld + kb + 1];
  }
  return a;
}
// A fragment whose 16 K-values come from one 32-half row of a [*, 32] table (per-lane row).
__device__ __forceinline__ v16h loadA_row(const _Float16* T, int row, int lane) {
  int hf = lane >> 4;
  v16h a;
#pragma unroll
  for (int v = 0; v < 8; ++v) {
    int kb = ((v >> 2) << 4) + (hf << 3) + ((v & 3) << 1);
    a[2 * v]     = T[row * 32 + kb];
    a[2 * v + 1] = T[row * 32 + kb + 1];
  }
  return a;
}
// B: 32x16 f16, element (k,n) = W[n0+n][k0+k], W stored [out][in_pad] in global (L2-hot).
__device__ __forceinline__ v16h loadBg(const _Float16* __restrict__ W, int ldw, int n0, int k0, int lane) {
  int n = lane & 15, hf = lane >> 4;
  v16h b;
#pragma unroll
  for (int v = 0; v < 8; ++v) {
    int k = k0 + ((v >> 2) << 4) + (hf << 3) + ((v & 3) << 1);
    b[2 * v]     = W[(n0 + n) * ldw + k];
    b[2 * v + 1] = W[(n0 + n) * ldw + k + 1];
  }
  return b;
}

__device__ __forceinline__ v8f splat8(float v) {
  v8f r;
#pragma unroll
  for (int i = 0; i < 8; ++i) r[i] = v;
  return r;
}

// Convert two f32 values to packed f16 (one v_cvt_pk_f16_f32), relu with one
// v_pk_max_num_f16, store with ONE ds_store_b32 (destination is LDS-contiguous
// thanks to the permuted hidden-layer storage layout).
__device__ __forceinline__ void store_relu_pair2(float a, float b, v2h* p) {
  v2h h;
  h[0] = (_Float16)a;
  h[1] = (_Float16)b;
  v2h z = {(_Float16)0.f, (_Float16)0.f};
  h = __builtin_elementwise_max(h, z);
  *p = h;
}

// Per-stage weights held in VGPRs across the whole tile loop.
struct WF {
  v16h  B1[2][4];   // layer1: [kt][nt]
  v16h  B2[2][2];   // layer2
  v16h  B3;         // layer3
  float b1[4], b2[2], b3;   // per-lane bias value for this lane's output column
};

__device__ __forceinline__ void preload(WF& f, const _Float16* __restrict__ wsH,
                                        const float* __restrict__ wsB,
                                        int ow1, int ow2, int ow3,
                                        int ob1, int ob2, int ob3, int lane) {
#pragma unroll
  for (int kt = 0; kt < 2; ++kt)
#pragma unroll
    for (int nt = 0; nt < 4; ++nt)
      f.B1[kt][nt] = loadBg(wsH + ow1, 64, nt * 16, kt * 32, lane);
#pragma unroll
  for (int kt = 0; kt < 2; ++kt)
#pragma unroll
    for (int nt = 0; nt < 2; ++nt)
      f.B2[kt][nt] = loadBg(wsH + ow2, 64, nt * 16, kt * 32, lane);
  f.B3 = loadBg(wsH + ow3, 32, 0, 0, lane);
  int n = lane & 15;
#pragma unroll
  for (int nt = 0; nt < 4; ++nt) f.b1[nt] = wsB[ob1 + nt * 16 + n];
#pragma unroll
  for (int nt = 0; nt < 2; ++nt) f.b2[nt] = wsB[ob2 + nt * 16 + n];
  f.b3 = wsB[ob3 + n];
}

#define WMMA(A, Bf, C) __builtin_amdgcn_wmma_f32_16x16x32_f16(false, (A), false, (Bf), (short)0, (C), false, false)

// Full 3-layer MLP on one 16-row tile. a0/a1 are the two K=32 input fragments.
// Biases folded into the WMMA C-operand; relu/cvt packed; hidden activations
// stored in K-permuted layouts matched by permuted weights.
// h1: 16x64 staging (col = n*4+nt), h2: 16x32 staging (col = n*2+nt).
__device__ __forceinline__ v8f mlp_tile(const WF& f, v16h a0, v16h a1,
                                        _Float16* h1, _Float16* h2, int lane) {
  const int n = lane & 15, hf = lane >> 4;
  // ---- layer 1 (K=64, N=64) : 8 WMMAs with resident weights, bias in C ----
  v8f c[4] = {splat8(f.b1[0]), splat8(f.b1[1]), splat8(f.b1[2]), splat8(f.b1[3])};
#pragma unroll
  for (int nt = 0; nt < 4; ++nt) c[nt] = WMMA(a0, f.B1[0][nt], c[nt]);
#pragma unroll
  for (int nt = 0; nt < 4; ++nt) c[nt] = WMMA(a1, f.B1[1][nt], c[nt]);
#pragma unroll
  for (int np = 0; np < 2; ++np) {     // fragment pairs (0,1) and (2,3)
#pragma unroll
    for (int v = 0; v < 8; ++v) {
      store_relu_pair2(c[2 * np][v], c[2 * np + 1][v],
                       (v2h*)(h1 + (v + (hf << 3)) * 64 + (n << 2) + 2 * np));
    }
  }
  // ---- layer 2 (K=64, N=32) : 4 WMMAs (weights pre-permuted to h1 layout) ----
  v16h a20 = loadA(h1, 64, 0, lane);
  v16h a21 = loadA(h1, 64, 32, lane);
  v8f d[2] = {splat8(f.b2[0]), splat8(f.b2[1])};
#pragma unroll
  for (int nt = 0; nt < 2; ++nt) d[nt] = WMMA(a20, f.B2[0][nt], d[nt]);
#pragma unroll
  for (int nt = 0; nt < 2; ++nt) d[nt] = WMMA(a21, f.B2[1][nt], d[nt]);
#pragma unroll
  for (int v = 0; v < 8; ++v) {
    store_relu_pair2(d[0][v], d[1][v],
                     (v2h*)(h2 + (v + (hf << 3)) * 32 + (n << 1)));
  }
  // ---- layer 3 (K=32, N=16) : 1 WMMA (weights pre-permuted to h2 layout) ----
  v16h a3 = loadA(h2, 32, 0, lane);
  v8f e = WMMA(a3, f.B3, splat8(f.b3));
#pragma unroll
  for (int v = 0; v < 8; ++v) {
    e[v] = __builtin_fmaxf(e[v], 0.f);
  }
  return e;
}

// ---------------- fused per-batch kernel: one workgroup (8 wave32) per batch ----------
__global__ __launch_bounds__(256)
void graphnet_fused(const float* __restrict__ x, const float* __restrict__ y,
                    const _Float16* __restrict__ wsH, const float* __restrict__ wsB,
                    const float* __restrict__ fcf_w, const float* __restrict__ fcf_b,
                    float* __restrict__ out) {
  __shared__ _Float16 sXT[NN * 32];   // x[b] transposed: [node][feat], feat padded to 32
  __shared__ _Float16 sYT[NV * 32];   // y[b] transposed: [variant][station], padded to 32
  __shared__ float    sEbar[NN * DE];
  __shared__ float    sEsv[NN * DE];
  __shared__ float    sNsum[8];
  __shared__ _Float16 sH[8][1536];    // per-wave staging: h1 (16x64) + h2 (16x32)

  const int b    = blockIdx.x;
  const int tid  = threadIdx.x;
  const int lane = tid & 31;
  const int wave = tid >> 5;
  _Float16* h1 = sH[wave];
  _Float16* h2 = sH[wave] + 1024;

  // ---- stage in x/y (transposed, padded); zero accumulators ----
  for (int i = tid; i < NN * 32; i += 256) {
    int node = i >> 5, p = i & 31;
    sXT[i] = (p < PP) ? (_Float16)x[(size_t)b * PP * NN + p * NN + node] : (_Float16)0.f;
  }
  for (int i = tid; i < NV * 32; i += 256) {
    int v = i >> 5, s = i & 31;
    sYT[i] = (s < SS) ? (_Float16)y[(size_t)b * SS * NV + s * NV + v] : (_Float16)0.f;
  }
  for (int i = tid; i < NN * DE; i += 256) { sEbar[i] = 0.f; sEsv[i] = 0.f; }
  if (tid < 8) sNsum[tid] = 0.f;
  __syncthreads();

  const int m = lane & 15, hf = lane >> 4;

  // ---- stage 1: pair-MLP + Ebar reduction (3540 rows, 222 tiles over 8 waves) ----
  {
    WF f;
    preload(f, wsH, wsB, OW1, OW2, OW3, OB1, OB2, OB3, lane);
    for (int t = wave; t * 16 < NPAIR; t += 8) {
      int q  = t * 16 + m;
      int qc = q < NPAIR ? q : 0;
      int r  = qc / (NN - 1);
      int s0 = qc - r * (NN - 1);
      int s  = s0 + (s0 >= r ? 1 : 0);
      v16h a0 = loadA_row(sXT, r, lane);   // cols 0..31  = x[:, r]
      v16h a1 = loadA_row(sXT, s, lane);   // cols 32..63 = x[:, s]
      v8f e = mlp_tile(f, a0, a1, h1, h2, lane);
      if ((lane & 15) < DE) {
        int n  = lane & 15;
        int qb = t * 16 + (hf << 3);
        int rb = qb / (NN - 1);
        int sb = qb - rb * (NN - 1);
#pragma unroll
        for (int v = 0; v < 8; ++v) {
          if (qb + v < NPAIR) atomicAdd(&sEbar[rb * DE + n], e[v]);
          if (++sb >= NN - 1) { sb = 0; ++rb; }   // incremental div by 59
        }
      }
    }
  }

  // ---- stage 2: (node,variant)-MLP + Ebar_sv reduction (300 rows, 19 tiles) ----
  {
    WF f;
    preload(f, wsH, wsB, OW1s, OW2s, OW3s, OB1s, OB2s, OB3s, lane);
    for (int t = wave; t * 16 < NSV; t += 8) {
      int j  = t * 16 + m;
      int jc = j < NSV ? j : 0;
      int node = jc / NV, vv = jc - node * NV;
      v16h a0 = loadA_row(sXT, node, lane);  // cols 0..31  = x[:, node]
      v16h a1 = loadA_row(sYT, vv, lane);    // cols 32..63 = y[:, vv] (padded)
      v8f e = mlp_tile(f, a0, a1, h1, h2, lane);
      if ((lane & 15) < DE) {
        int n  = lane & 15;
        int jb = t * 16 + (hf << 3);
        int nb = jb / NV;
        int cb = jb - nb * NV;
#pragma unroll
        for (int v = 0; v < 8; ++v) {
          if (jb + v < NSV) atomicAdd(&sEsv[nb * DE + n], e[v]);
          if (++cb >= NV) { cb = 0; ++nb; }       // incremental div by 5
        }
      }
    }
  }
  __syncthreads();

  // ---- stage 3: node-MLP on C = [x^T | Ebar | Ebar_sv] (60 rows, 4 tiles, waves 0..3) ----
  if (wave < 4) {
    WF f;
    preload(f, wsH, wsB, OWo1, OWo2, OWo3, OBo1, OBo2, OBo3, lane);
    int t = wave;
    int node = t * 16 + m;
    int nc = node < NN ? node : 0;
    v16h a0 = loadA_row(sXT, nc, lane);       // cols 0..31 = x[:, node]
    v16h a1;                                   // cols 32..41 = Ebar | Esv, rest 0
#pragma unroll
    for (int v = 0; v < 8; ++v) {
      int kb = ((v >> 2) << 4) + (hf << 3) + ((v & 3) << 1);
#pragma unroll
      for (int u = 0; u < 2; ++u) {
        int k = kb + u;
        float val = 0.f;
        if (k < DE)           val = sEbar[nc * DE + k];
        else if (k < 2 * DE)  val = sEsv[nc * DE + (k - DE)];
        a1[2 * v + u] = (_Float16)val;
      }
    }
    v8f o = mlp_tile(f, a0, a1, h1, h2, lane);
    if ((lane & 15) < DO_) {
      int n = lane & 15;
#pragma unroll
      for (int v = 0; v < 8; ++v) {
        int nn = t * 16 + v + (hf << 3);
        if (nn < NN) atomicAdd(&sNsum[n], o[v]);
      }
    }
  }
  __syncthreads();

  // ---- head: out[b] = relu(Nsum @ fcf_w^T + fcf_b) ----
  if (tid < NT_) {
    float acc = fcf_b[tid];
#pragma unroll
    for (int n = 0; n < DO_; ++n) acc += sNsum[n] * fcf_w[tid * DO_ + n];
    out[(size_t)b * NT_ + tid] = __builtin_fmaxf(acc, 0.f);
  }
}

extern "C" void kernel_launch(void* const* d_in, const int* in_sizes, int n_in,
                              void* d_out, int out_size, void* d_ws, size_t ws_size,
                              hipStream_t stream) {
  const float* x      = (const float*)d_in[0];
  const float* y      = (const float*)d_in[1];
  const float* fr1_w  = (const float*)d_in[2];  const float* fr1_b  = (const float*)d_in[3];
  const float* fr2_w  = (const float*)d_in[4];  const float* fr2_b  = (const float*)d_in[5];
  const float* fr3_w  = (const float*)d_in[6];  const float* fr3_b  = (const float*)d_in[7];
  const float* fr1s_w = (const float*)d_in[8];  const float* fr1s_b = (const float*)d_in[9];
  const float* fr2s_w = (const float*)d_in[10]; const float* fr2s_b = (const float*)d_in[11];
  const float* fr3s_w = (const float*)d_in[12]; const float* fr3s_b = (const float*)d_in[13];
  const float* fo1_w  = (const float*)d_in[14]; const float* fo1_b  = (const float*)d_in[15];
  const float* fo2_w  = (const float*)d_in[16]; const float* fo2_b  = (const float*)d_in[17];
  const float* fo3_w  = (const float*)d_in[18]; const float* fo3_b  = (const float*)d_in[19];
  const float* fcf_w  = (const float*)d_in[20]; const float* fcf_b  = (const float*)d_in[21];

  _Float16* wsH = (_Float16*)d_ws;
  float*    wsB = (float*)((char*)d_ws + WSB_BYTE_OFF);

  prep_weights<<<64, 256, 0, stream>>>(fr1_w, fr1_b, fr2_w, fr2_b, fr3_w, fr3_b,
                                       fr1s_w, fr1s_b, fr2s_w, fr2s_b, fr3s_w, fr3s_b,
                                       fo1_w, fo1_b, fo2_w, fo2_b, fo3_w, fo3_b,
                                       wsH, wsB);
  graphnet_fused<<<BB, 256, 0, stream>>>(x, y, wsH, wsB, fcf_w, fcf_b, (float*)d_out);
}
